// VanillaTransformer_33621003994012
// MI455X (gfx1250) — matrix operand order repaired and checked by reading, not compiled, not run on hardware
//
#include <hip/hip_runtime.h>
#include <hip/hip_bf16.h>
#include <math.h>

// ---------------------------------------------------------------------------
// CDNA5 (gfx1250) wave32 WMMA transformer forward.
//   B=2, S=2048, D=768, H=12, HD=64, L=6, V=32000
//   GEMMs: bf16 WMMA, tiles staged by the Tensor Data Mover (TDM) with
//   double-buffered LDS; attention: flash-style online softmax, all WMMA.
// ---------------------------------------------------------------------------

#define USE_TDM 1

typedef __bf16 bf16;
typedef __bf16 v16bf  __attribute__((ext_vector_type(16)));
typedef __bf16 bf16x8 __attribute__((ext_vector_type(8)));
typedef float  v8f    __attribute__((ext_vector_type(8)));
typedef unsigned int u32x4 __attribute__((ext_vector_type(4)));
typedef int    i32x4  __attribute__((ext_vector_type(4)));
typedef int    i32x8  __attribute__((ext_vector_type(8)));

#define WMMA_BF16(a, b, c) \
  __builtin_amdgcn_wmma_f32_16x16x32_bf16(false, (a), false, (b), (short)0, (c), false, false)

// ---------------------------------------------------------------------------
// TDM: DMA a 128-row x 32-col bf16 tile (row stride K elements) from global
// into LDS with 16B row padding (LDS row stride = 40 bf16 = 80B).
// D# bit layout per CDNA5 ISA:
//   group0: [1:0]=count=1, [63:32]=lds_addr, [120:64]=global_addr, [127:126]=2
//   group1: data_size=1(2B), pad_enable, pad_interval=3 (64B), pad_amount=3
//           (4 DWORDs = 16B), tensor_dim0=K, tensor_dim1=rows,
//           tile_dim0=32, tile_dim1=128, tensor_dim0_stride=K
// ---------------------------------------------------------------------------
#if USE_TDM
__device__ __forceinline__ void tdm_load_tile(const bf16* gptr, unsigned lds_addr,
                                              int K, int rows) {
  unsigned long long ga = (unsigned long long)(size_t)gptr;
  u32x4 g0;
  g0[0] = 1u;                                                  // count=1
  g0[1] = lds_addr;                                            // LDS byte addr
  g0[2] = (unsigned)(ga & 0xFFFFFFFFu);                        // global[31:0]
  g0[3] = (unsigned)((ga >> 32) & 0x01FFFFFFu) | 0x80000000u;  // global[56:32]|type=2
  i32x8 g1;
  g1[0] = (int)0x06D10000u;  // data_size=2B, pad_en, pad_interval=64B, pad=16B
  g1[1] = (int)(((unsigned)K & 0xFFFFu) << 16);                // tensor_dim0 lo
  g1[2] = (int)(((unsigned)K >> 16) | (((unsigned)rows & 0xFFFFu) << 16));
  g1[3] = (int)(((unsigned)rows >> 16) | (32u << 16));         // tile_dim0=32
  g1[4] = 128;                                                 // tile_dim1=128
  g1[5] = K;                                                   // dim0 stride lo
  g1[6] = 0;
  g1[7] = 0;
  i32x4 z4 = {0, 0, 0, 0};
#if __clang_major__ >= 23
  i32x8 z8 = {0, 0, 0, 0, 0, 0, 0, 0};
  __builtin_amdgcn_tensor_load_to_lds(g0, g1, z4, z4, z8, 0);
#else
  __builtin_amdgcn_tensor_load_to_lds(g0, g1, z4, z4, 0);
#endif
}
#endif

// ---------------------------------------------------------------------------
// f32 -> bf16 conversion (weights / activations)
// ---------------------------------------------------------------------------
__global__ void cvt_kernel(const float* __restrict__ in, bf16* __restrict__ out, int n) {
  int i = blockIdx.x * blockDim.x + threadIdx.x;
  int stride = gridDim.x * blockDim.x;
  for (; i < n; i += stride) out[i] = (bf16)in[i];
}

// ---------------------------------------------------------------------------
// x[b,s,:] = tok_emb[idx[b,s]] + pos_enc[s]   (one block per row, 768 = 3*256)
// ---------------------------------------------------------------------------
__global__ __launch_bounds__(256)
void embed_kernel(const int* __restrict__ idx, const float* __restrict__ tok,
                  const float* __restrict__ pos, float* __restrict__ x) {
  int row = blockIdx.x;           // 0..4095  (b*2048 + s)
  int s = row & 2047;
  int token = idx[row];
#pragma unroll
  for (int i = 0; i < 3; ++i) {
    int c = threadIdx.x + i * 256;
    x[(size_t)row * 768 + c] = tok[(size_t)token * 768 + c] + pos[(size_t)s * 768 + c];
  }
}

// ---------------------------------------------------------------------------
// LayerNorm row kernel: out_bf16 = (x - mu) * rsqrt(var+eps) * g + b
// ---------------------------------------------------------------------------
__global__ __launch_bounds__(256)
void ln_kernel(const float* __restrict__ x, const float* __restrict__ g,
               const float* __restrict__ bb, bf16* __restrict__ out) {
  __shared__ float red[16];
  int row = blockIdx.x;
  const float* xr = x + (size_t)row * 768;
  float s = 0.f, ss = 0.f;
#pragma unroll
  for (int i = 0; i < 3; ++i) {
    float v = xr[threadIdx.x + i * 256];
    s += v; ss += v * v;
  }
#pragma unroll
  for (int off = 1; off < 32; off <<= 1) {
    s  += __shfl_xor(s,  off, 32);
    ss += __shfl_xor(ss, off, 32);
  }
  int w = threadIdx.x >> 5, lane = threadIdx.x & 31;
  if (lane == 0) { red[w] = s; red[8 + w] = ss; }
  __syncthreads();
  float ts = 0.f, tss = 0.f;
#pragma unroll
  for (int i = 0; i < 8; ++i) { ts += red[i]; tss += red[8 + i]; }
  float mean = ts * (1.0f / 768.0f);
  float var  = tss * (1.0f / 768.0f) - mean * mean;
  float rstd = rsqrtf(var + 1e-5f);
#pragma unroll
  for (int i = 0; i < 3; ++i) {
    int c = threadIdx.x + i * 256;
    float v = (xr[c] - mean) * rstd * g[c] + bb[c];
    out[(size_t)row * 768 + c] = (bf16)v;
  }
}

// ---------------------------------------------------------------------------
// Tiled bf16 WMMA GEMM:  C[M,N] = A[M,K] @ W[N,K]^T (+bias) (+epilogue)
//   Block tile 128x128, 8 wave32 waves (4x2), wave tile 32x64, k-step 32.
//   Tiles staged by TDM (wave 0 issues tensor_load_to_lds, TENSORcnt-tracked)
//   into double-buffered LDS; compute of tile i overlaps DMA of tile i+1.
//   EPI: 0 = f32 out (+bias)          (decode head)
//        1 = f32 out = resid + bias+C (residual projections)
//        2 = bf16 out = gelu(bias+C)  (FC1)
//        3 = bf16 out = bias+C        (QKV)
// ---------------------------------------------------------------------------
template <int EPI>
__global__ __launch_bounds__(256)
void gemm_bf16_kernel(const bf16* __restrict__ A, const bf16* __restrict__ W,
                      const float* __restrict__ bias, const float* __restrict__ resid,
                      float* __restrict__ outF, bf16* __restrict__ outB,
                      int M, int N, int K) {
  __shared__ __align__(16) bf16 As[2][128][40];   // 32 K + 8 pad, 16B rows
  __shared__ __align__(16) bf16 Bs[2][128][40];

  const int t = threadIdx.x;
  const int lane = t & 31;
  const int wid = t >> 5;
  const int wm = wid >> 1;               // 0..3 -> 32-row slab
  const int wn = wid & 1;                // 0..1 -> 64-col slab
  const int m0 = blockIdx.y * 128;
  const int n0 = blockIdx.x * 128;
  const int nk = K >> 5;

#if USE_TDM
  if (wid == 0) {                        // prologue: DMA tile 0 into buffer 0
    tdm_load_tile(A + (size_t)m0 * K, (unsigned)(size_t)&As[0][0][0], K, M);
    tdm_load_tile(W + (size_t)n0 * K, (unsigned)(size_t)&Bs[0][0][0], K, N);
  }
#endif

  v8f acc[2][4] = {};

  for (int i = 0; i < nk; ++i) {
    const int buf = i & 1;
    const int k0 = i << 5;
#if USE_TDM
    if (wid == 0) __builtin_amdgcn_s_wait_tensorcnt(0);   // tile i landed
    __syncthreads();                                      // visible to all waves
    if (wid == 0 && i + 1 < nk) {                         // start tile i+1
      tdm_load_tile(A + (size_t)m0 * K + k0 + 32,
                    (unsigned)(size_t)&As[buf ^ 1][0][0], K, M);
      tdm_load_tile(W + (size_t)n0 * K + k0 + 32,
                    (unsigned)(size_t)&Bs[buf ^ 1][0][0], K, N);
    }
#else
    __syncthreads();
#pragma unroll
    for (int p = 0; p < 2; ++p) {
      int idx = p * 256 + t;
      int r = idx >> 2, c = (idx & 3) * 8;
      *(bf16x8*)&As[buf][r][c] = *(const bf16x8*)(A + (size_t)(m0 + r) * K + k0 + c);
      *(bf16x8*)&Bs[buf][r][c] = *(const bf16x8*)(W + (size_t)(n0 + r) * K + k0 + c);
    }
    __syncthreads();
#endif

    // ---- fragments in CDNA5 WMMA VGPR layouts ----
    const int ka = (lane & 16) ? 8 : 0;    // A: lanes16-31 hold K=8..15 / 24..31
    const int kb = (lane & 16) ? 8 : 0;    // B: lanes16-31 hold K=8..15 / 24..31
    v16bf af[2], bfr[4];
#pragma unroll
    for (int mt = 0; mt < 2; ++mt) {
      int r = wm * 32 + mt * 16 + (lane & 15);
      union { v16bf v; bf16x8 h[2]; } u;
      u.h[0] = *(const bf16x8*)&As[buf][r][ka];
      u.h[1] = *(const bf16x8*)&As[buf][r][ka + 16];
      af[mt] = u.v;
    }
#pragma unroll
    for (int nt = 0; nt < 4; ++nt) {
      int r = wn * 64 + nt * 16 + (lane & 15);
      union { v16bf v; bf16x8 h[2]; } u;
      u.h[0] = *(const bf16x8*)&Bs[buf][r][kb];
      u.h[1] = *(const bf16x8*)&Bs[buf][r][kb + 16];
      bfr[nt] = u.v;
    }
#pragma unroll
    for (int mt = 0; mt < 2; ++mt)
#pragma unroll
      for (int nt = 0; nt < 4; ++nt)
        acc[mt][nt] = WMMA_BF16(af[mt], bfr[nt], acc[mt][nt]);
  }

  // ---- epilogue (C layout: VGPR r -> row r / r+8 by lane half, col = lane&15)
#pragma unroll
  for (int mt = 0; mt < 2; ++mt)
#pragma unroll
    for (int nt = 0; nt < 4; ++nt)
#pragma unroll
      for (int r = 0; r < 8; ++r) {
        int row = m0 + wm * 32 + mt * 16 + ((lane & 16) ? r + 8 : r);
        int col = n0 + wn * 64 + nt * 16 + (lane & 15);
        float v = acc[mt][nt][r];
        if (bias) v += bias[col];
        size_t o = (size_t)row * N + col;
        if (EPI == 0) {
          outF[o] = v;
        } else if (EPI == 1) {
          outF[o] = resid[o] + v;
        } else if (EPI == 2) {
          float c3 = v * v * v;
          float gl = 0.5f * v * (1.0f + tanhf(0.7978845608028654f * (v + 0.044715f * c3)));
          outB[o] = (bf16)gl;
        } else {
          outB[o] = (bf16)v;
        }
      }
}

// ---------------------------------------------------------------------------
// Flash attention (no mask): per (q-block of 128, head, batch).
//   qkv layout: [b, s, 3*768] rows; q at +0, k at +768, v at +1536, head h at +h*64.
//   8 waves: wave w owns 16 q-rows.  K-blocks of 64 streamed through LDS.
//   QK^T and P@V both via v_wmma_f32_16x16x32_bf16 (HD=64 -> 2 WMMA per tile).
// ---------------------------------------------------------------------------
__global__ __launch_bounds__(256)
void flash_attn_kernel(const bf16* __restrict__ qkv, bf16* __restrict__ y) {
  __shared__ __align__(16) bf16 Qs[128][72];
  __shared__ __align__(16) bf16 Ks[64][72];
  __shared__ __align__(16) bf16 Vt[64][72];      // V transposed: Vt[d][k]
  __shared__ __align__(16) bf16 Ps[8][16][72];   // per-wave P tile

  const int t = threadIdx.x, lane = t & 31, w = t >> 5;
  const int q0 = blockIdx.x * 128;
  const int h = blockIdx.y, b = blockIdx.z;
  const size_t rs = 2304;
  const bf16* base = qkv + (size_t)(b * 2048) * rs;

  // ---- load Q tile (128 x 64) ----
#pragma unroll
  for (int p = 0; p < 4; ++p) {
    int idx = p * 256 + t;
    int r = idx >> 3, c = (idx & 7) * 8;
    *(bf16x8*)&Qs[r][c] = *(const bf16x8*)(base + (size_t)(q0 + r) * rs + h * 64 + c);
  }

  v8f oacc[4] = {};
  float mrow[8], lrow[8];
#pragma unroll
  for (int r = 0; r < 8; ++r) { mrow[r] = -3.0e38f; lrow[r] = 0.0f; }

  const int ka = (lane & 16) ? 8 : 0;
  const int kb = (lane & 16) ? 8 : 0;

  for (int k0 = 0; k0 < 2048; k0 += 64) {
    __syncthreads();                       // protect Ks/Vt reuse
    // ---- K block (64 x 64, row-major = ready-made B fragment layout) ----
#pragma unroll
    for (int p = 0; p < 2; ++p) {
      int idx = p * 256 + t;
      int r = idx >> 3, c = (idx & 7) * 8;
      *(bf16x8*)&Ks[r][c] = *(const bf16x8*)(base + (size_t)(k0 + r) * rs + 768 + h * 64 + c);
    }
    // ---- V block, transposed into Vt[d][k] ----
    {
      int kk = t >> 2, d0 = (t & 3) * 16;
      const bf16* src = base + (size_t)(k0 + kk) * rs + 1536 + h * 64 + d0;
      bf16x8 v0 = *(const bf16x8*)src;
      bf16x8 v1 = *(const bf16x8*)(src + 8);
#pragma unroll
      for (int j = 0; j < 8; ++j) { Vt[d0 + j][kk] = v0[j]; Vt[d0 + 8 + j][kk] = v1[j]; }
    }
    __syncthreads();

    // ---- S = Q K^T  (wave tile 16 x 64) ----
    int qr = w * 16 + (lane & 15);
    v16bf a0, a1;
    {
      union { v16bf v; bf16x8 hh[2]; } u;
      u.hh[0] = *(const bf16x8*)&Qs[qr][ka];      u.hh[1] = *(const bf16x8*)&Qs[qr][ka + 16];
      a0 = u.v;
      u.hh[0] = *(const bf16x8*)&Qs[qr][32 + ka]; u.hh[1] = *(const bf16x8*)&Qs[qr][32 + ka + 16];
      a1 = u.v;
    }
    v8f sacc[4] = {};
#pragma unroll
    for (int nt = 0; nt < 4; ++nt) {
      int kr = nt * 16 + (lane & 15);
      union { v16bf v; bf16x8 hh[2]; } u;
      u.hh[0] = *(const bf16x8*)&Ks[kr][kb];      u.hh[1] = *(const bf16x8*)&Ks[kr][kb + 16];
      v16bf b0 = u.v;
      u.hh[0] = *(const bf16x8*)&Ks[kr][32 + kb]; u.hh[1] = *(const bf16x8*)&Ks[kr][32 + kb + 16];
      v16bf b1 = u.v;
      sacc[nt] = WMMA_BF16(a0, b0, sacc[nt]);
      sacc[nt] = WMMA_BF16(a1, b1, sacc[nt]);
    }

    // ---- online softmax (rows live in 16-lane halves; reduce with shfl_xor) --
#pragma unroll
    for (int r = 0; r < 8; ++r) {
      float rmax = -3.0e38f;
#pragma unroll
      for (int nt = 0; nt < 4; ++nt) rmax = fmaxf(rmax, sacc[nt][r] * 0.125f);
#pragma unroll
      for (int off = 1; off < 16; off <<= 1)
        rmax = fmaxf(rmax, __shfl_xor(rmax, off, 32));
      float mnew = fmaxf(mrow[r], rmax);
      float alpha = __expf(mrow[r] - mnew);
      float rsum = 0.0f;
#pragma unroll
      for (int nt = 0; nt < 4; ++nt) {
        float pe = __expf(sacc[nt][r] * 0.125f - mnew);
        rsum += pe;
        int prow = (lane & 16) ? r + 8 : r;
        int pcol = nt * 16 + (lane & 15);
        Ps[w][prow][pcol] = (bf16)pe;      // per-wave buffer, wave-order via DScnt
      }
#pragma unroll
      for (int off = 1; off < 16; off <<= 1)
        rsum += __shfl_xor(rsum, off, 32);
      lrow[r] = lrow[r] * alpha + rsum;
      mrow[r] = mnew;
#pragma unroll
      for (int nt = 0; nt < 4; ++nt) oacc[nt][r] *= alpha;
    }

    // ---- O += P @ V ----
    v16bf p0, p1;
    {
      int pr = lane & 15;
      union { v16bf v; bf16x8 hh[2]; } u;
      u.hh[0] = *(const bf16x8*)&Ps[w][pr][ka];      u.hh[1] = *(const bf16x8*)&Ps[w][pr][ka + 16];
      p0 = u.v;
      u.hh[0] = *(const bf16x8*)&Ps[w][pr][32 + ka]; u.hh[1] = *(const bf16x8*)&Ps[w][pr][32 + ka + 16];
      p1 = u.v;
    }
#pragma unroll
    for (int nt = 0; nt < 4; ++nt) {
      int vr = nt * 16 + (lane & 15);
      union { v16bf v; bf16x8 hh[2]; } u;
      u.hh[0] = *(const bf16x8*)&Vt[vr][kb];      u.hh[1] = *(const bf16x8*)&Vt[vr][kb + 16];
      v16bf b0 = u.v;
      u.hh[0] = *(const bf16x8*)&Vt[vr][32 + kb]; u.hh[1] = *(const bf16x8*)&Vt[vr][32 + kb + 16];
      v16bf b1 = u.v;
      oacc[nt] = WMMA_BF16(p0, b0, oacc[nt]);
      oacc[nt] = WMMA_BF16(p1, b1, oacc[nt]);
    }
  }

  // ---- write y[b, q, h*64 + d] = O / l ----
#pragma unroll
  for (int nt = 0; nt < 4; ++nt)
#pragma unroll
    for (int r = 0; r < 8; ++r) {
      int q = q0 + w * 16 + ((lane & 16) ? r + 8 : r);
      int d = nt * 16 + (lane & 15);
      float v = oacc[nt][r] / lrow[r];
      y[((size_t)(b * 2048 + q)) * 768 + h * 64 + d] = (bf16)v;
    }
}

// ---------------------------------------------------------------------------
// Orchestration
// ---------------------------------------------------------------------------
extern "C" void kernel_launch(void* const* d_in, const int* in_sizes, int n_in,
                              void* d_out, int out_size, void* d_ws, size_t ws_size,
                              hipStream_t stream) {
  const int*   idx    = (const int*)  d_in[0];
  const float* tok    = (const float*)d_in[1];
  const float* pos    = (const float*)d_in[2];
  const float* attn_w = (const float*)d_in[3];
  const float* attn_b = (const float*)d_in[4];
  const float* proj_w = (const float*)d_in[5];
  const float* proj_b = (const float*)d_in[6];
  const float* ln1_g  = (const float*)d_in[7];
  const float* ln1_b  = (const float*)d_in[8];
  const float* ln2_g  = (const float*)d_in[9];
  const float* ln2_b  = (const float*)d_in[10];
  const float* fc1_w  = (const float*)d_in[11];
  const float* fc1_b  = (const float*)d_in[12];
  const float* fc2_w  = (const float*)d_in[13];
  const float* fc2_b  = (const float*)d_in[14];
  const float* dec_w  = (const float*)d_in[15];
  float* out = (float*)d_out;

  char* ws = (char*)d_ws;
  size_t off = 0;
  auto alloc = [&](size_t bytes) {
    void* p = ws + off;
    off += (bytes + 255) & ~(size_t)255;
    return p;
  };

  bf16*  wqkv = (bf16*)alloc((size_t)6 * 2304 * 768 * 2);
  bf16*  wprj = (bf16*)alloc((size_t)6 * 768 * 768 * 2);
  bf16*  wfc1 = (bf16*)alloc((size_t)6 * 3072 * 768 * 2);
  bf16*  wfc2 = (bf16*)alloc((size_t)6 * 768 * 3072 * 2);
  bf16*  wdec = (bf16*)alloc((size_t)32000 * 768 * 2);
  float* x    = (float*)alloc((size_t)4096 * 768 * 4);
  bf16*  hb   = (bf16*)alloc((size_t)4096 * 768 * 2);
  bf16*  qkvb = (bf16*)alloc((size_t)4096 * 2304 * 2);
  bf16*  yb   = (bf16*)alloc((size_t)4096 * 768 * 2);
  bf16*  h2b  = (bf16*)alloc((size_t)4096 * 3072 * 2);
  bf16*  xb   = (bf16*)alloc((size_t)4096 * 768 * 2);

  dim3 blk(256);

  cvt_kernel<<<2048, blk, 0, stream>>>(attn_w, wqkv, 6 * 2304 * 768);
  cvt_kernel<<<2048, blk, 0, stream>>>(proj_w, wprj, 6 * 768 * 768);
  cvt_kernel<<<2048, blk, 0, stream>>>(fc1_w,  wfc1, 6 * 3072 * 768);
  cvt_kernel<<<2048, blk, 0, stream>>>(fc2_w,  wfc2, 6 * 768 * 3072);
  cvt_kernel<<<2048, blk, 0, stream>>>(dec_w,  wdec, 32000 * 768);
  embed_kernel<<<4096, blk, 0, stream>>>(idx, tok, pos, x);

  for (int l = 0; l < 6; ++l) {
    ln_kernel<<<4096, blk, 0, stream>>>(x, ln1_g + l * 768, ln1_b + l * 768, hb);
    gemm_bf16_kernel<3><<<dim3(2304 / 128, 4096 / 128), blk, 0, stream>>>(
        hb, wqkv + (size_t)l * 2304 * 768, attn_b + l * 2304,
        nullptr, nullptr, qkvb, 4096, 2304, 768);
    flash_attn_kernel<<<dim3(2048 / 128, 12, 2), blk, 0, stream>>>(qkvb, yb);
    gemm_bf16_kernel<1><<<dim3(768 / 128, 4096 / 128), blk, 0, stream>>>(
        yb, wprj + (size_t)l * 768 * 768, proj_b + l * 768,
        x, x, nullptr, 4096, 768, 768);
    ln_kernel<<<4096, blk, 0, stream>>>(x, ln2_g + l * 768, ln2_b + l * 768, hb);
    gemm_bf16_kernel<2><<<dim3(3072 / 128, 4096 / 128), blk, 0, stream>>>(
        hb, wfc1 + (size_t)l * 3072 * 768, fc1_b + l * 3072,
        nullptr, nullptr, h2b, 4096, 3072, 768);
    gemm_bf16_kernel<1><<<dim3(768 / 128, 4096 / 128), blk, 0, stream>>>(
        h2b, wfc2 + (size_t)l * 768 * 3072, fc2_b + l * 768,
        x, x, nullptr, 4096, 768, 3072);
  }

  cvt_kernel<<<2048, blk, 0, stream>>>(x, xb, 4096 * 768);
  gemm_bf16_kernel<0><<<dim3(32000 / 128, 4096 / 128), blk, 0, stream>>>(
      xb, wdec, nullptr, nullptr, out, nullptr, 4096, 32000, 768);

  (void)in_sizes; (void)n_in; (void)out_size; (void)ws_size;
}
